// LLMAdapter_17051020165614
// MI455X (gfx1250) — matrix operand order repaired and checked by reading, not compiled, not run on hardware
//
#include <hip/hip_runtime.h>
#include <hip/hip_bf16.h>
#include <math.h>

// ---------------- model dims ----------------
#define B_   4
#define SQ_  1024
#define SKV_ 1024
#define SRC_ 4096
#define TGT_ 768
#define DM_  1024
#define NH_  16
#define HD_  64
#define DFF_ 4096
#define ROWS_ 4096            // B_*SQ_ == B_*SKV_

typedef __attribute__((ext_vector_type(16))) __bf16 v16bf;
typedef __attribute__((ext_vector_type(8)))  __bf16 v8bf;
typedef __attribute__((ext_vector_type(4)))  __bf16 v4bf;
typedef __attribute__((ext_vector_type(8)))  float  v8f;
typedef __attribute__((ext_vector_type(4)))  float  v4f;

// ---- CDNA5 async global->LDS copy (ASYNCcnt) ----
__device__ __forceinline__ void async_ld_b128(unsigned lds_off, const void* gaddr) {
    asm volatile("global_load_async_to_lds_b128 %0, %1, off"
                 :: "v"(lds_off), "v"(gaddr) : "memory");
}
__device__ __forceinline__ void wait_async0() {
    asm volatile("s_wait_asynccnt 0" ::: "memory");
}

// =====================================================================
// Tiled WMMA GEMM:  C[m,n] = act(scale * sum_k A[m,k]*B'[k,n] + bias[n]) + R[m,n]
//   BT=true : B stored (N,K) row-major (weights, K matrices)  -> B'[k,n]=B[n,k]
//   BT=false: B stored (K,N) row-major (V matrices)           -> B'[k,n]=B[k,n]
// 128x64 tile per 256-thread block (8 waves, 32x32 each); K-step 32.
// bf16 inputs (packed f32->bf16 convert), f32 accumulate.
// Software-pipelined: register staging of tile k+1 overlaps WMMA on tile k,
// double-buffered LDS, one barrier per K-step.
// Assumes M%128==0, N%64==0, K%32==0, lda/ldb%4==0, 16B-aligned rows.
// =====================================================================
#define LDSK 40   // padded bf16 row stride (80B = 5*16B: keeps b128/b64 LDS alignment)

template<bool BT, bool BIAS, bool GELU, bool RESID>
__global__ __launch_bounds__(256)
void gemm_wmma(const float* __restrict__ A, int lda, long sA,
               const float* __restrict__ B, int ldb, long sB,
               float* C, int ldc, long sC,
               const float* __restrict__ bias,
               const float* R,
               int M, int N, int K, float scale)
{
    __shared__ __bf16 Ash[2][128 * LDSK];
    __shared__ __bf16 Bsh[2][64 * LDSK];

    const int bz = blockIdx.z;
    A += (long)bz * sA;
    B += (long)bz * sB;
    C += (long)bz * sC;
    if (RESID) R += (long)bz * sC;

    const int m0 = blockIdx.y * 128;
    const int n0 = blockIdx.x * 64;
    const int tid  = threadIdx.x;
    const int lane = tid & 31;
    const int wave = tid >> 5;
    const int mw = (wave >> 1) * 32;       // wave's 32x32 sub-tile: mw in {0,32,64,96}
    const int nw = (wave & 1) * 32;        //                        nw in {0,32}
    const int g = lane >> 4;               // lane group (0/1)
    const int r = lane & 15;

    v4f aR[4], bR[2];                      // register-staged next tile (f32)

    auto load_tile = [&](int k0) {
        #pragma unroll
        for (int i = 0; i < 4; ++i) {                    // A: 128x32 = 1024 float4
            int linear = tid + i * 256;
            int row = linear >> 3, colq = (linear & 7) * 4;
            aR[i] = *(const v4f*)&A[(long)(m0 + row) * lda + (k0 + colq)];
        }
        if (BT) {
            #pragma unroll
            for (int i = 0; i < 2; ++i) {                // B: 64x32 = 512 float4
                int linear = tid + i * 256;
                int row = linear >> 3, colq = (linear & 7) * 4;
                bR[i] = *(const v4f*)&B[(long)(n0 + row) * ldb + (k0 + colq)];
            }
        } else {
            #pragma unroll
            for (int i = 0; i < 2; ++i) {                // B: 32(k) x 64(n), read along n
                int linear = tid + i * 256;
                int k = linear >> 4, nq = (linear & 15) * 4;
                bR[i] = *(const v4f*)&B[(long)(k0 + k) * ldb + (n0 + nq)];
            }
        }
    };
    auto store_tile = [&](int buf) {
        #pragma unroll
        for (int i = 0; i < 4; ++i) {
            int linear = tid + i * 256;
            int row = linear >> 3, colq = (linear & 7) * 4;
            v4bf h = { (__bf16)aR[i][0], (__bf16)aR[i][1], (__bf16)aR[i][2], (__bf16)aR[i][3] };
            *(v4bf*)&Ash[buf][row * LDSK + colq] = h;
        }
        if (BT) {
            #pragma unroll
            for (int i = 0; i < 2; ++i) {
                int linear = tid + i * 256;
                int row = linear >> 3, colq = (linear & 7) * 4;
                v4bf h = { (__bf16)bR[i][0], (__bf16)bR[i][1], (__bf16)bR[i][2], (__bf16)bR[i][3] };
                *(v4bf*)&Bsh[buf][row * LDSK + colq] = h;
            }
        } else {
            #pragma unroll
            for (int i = 0; i < 2; ++i) {
                int linear = tid + i * 256;
                int k = linear >> 4, nq = (linear & 15) * 4;
                #pragma unroll
                for (int j = 0; j < 4; ++j)
                    Bsh[buf][(nq + j) * LDSK + k] = (__bf16)bR[i][j];
            }
        }
    };

    v8f acc[2][2] = {};

    load_tile(0);
    store_tile(0);
    __syncthreads();

    int cur = 0;
    for (int k0 = 0; k0 < K; k0 += 32) {
        const bool has_next = (k0 + 32 < K);
        if (has_next) load_tile(k0 + 32);          // global loads overlap WMMA below
        if (k0 + 64 < K && tid < 128)              // prefetch 2 tiles ahead (global_prefetch_b8)
            __builtin_prefetch(&A[(long)(m0 + tid) * lda + (k0 + 64)], 0, 1);

        const __bf16* AshC = Ash[cur];
        const __bf16* BshC = Bsh[cur];
        // ---- fragments per documented wave32 layouts ----
        v16bf af[2], bfm[2];
        #pragma unroll
        for (int mi = 0; mi < 2; ++mi) {
            const __bf16* ap = &AshC[(mw + mi * 16 + r) * LDSK];
            v8bf lo = *(const v8bf*)(ap + g * 8);            // K = g*8 .. g*8+7
            v8bf hi = *(const v8bf*)(ap + 16 + g * 8);       // K = 16+g*8 ..
            af[mi] = __builtin_shufflevector(lo, hi, 0,1,2,3,4,5,6,7,8,9,10,11,12,13,14,15);
        }
        #pragma unroll
        for (int ni = 0; ni < 2; ++ni) {
            const __bf16* bp = &BshC[(nw + ni * 16 + r) * LDSK];
            v8bf lo = *(const v8bf*)(bp + g * 8);            // K = g*8 .. g*8+7
            v8bf hi = *(const v8bf*)(bp + 16 + g * 8);       // K = 16+g*8 ..
            bfm[ni] = __builtin_shufflevector(lo, hi, 0,1,2,3,4,5,6,7,8,9,10,11,12,13,14,15);
        }
        #pragma unroll
        for (int mi = 0; mi < 2; ++mi)
            #pragma unroll
            for (int ni = 0; ni < 2; ++ni)
                acc[mi][ni] = __builtin_amdgcn_wmma_f32_16x16x32_bf16(
                    false, af[mi], false, bfm[ni], (short)0, acc[mi][ni], false, false);

        if (has_next) store_tile(cur ^ 1);         // fill other buffer
        __syncthreads();                           // one barrier per K-step
        cur ^= 1;
    }

    // ---- epilogue: lane r = N col, vgpr e -> M = g*8+e; strength-reduced addresses ----
    #pragma unroll
    for (int mi = 0; mi < 2; ++mi)
        #pragma unroll
        for (int ni = 0; ni < 2; ++ni) {
            const int n = n0 + nw + ni * 16 + r;
            const float bv = BIAS ? bias[n] : 0.0f;
            long base = (long)(m0 + mw + mi * 16 + g * 8) * ldc + n;
            #pragma unroll
            for (int e = 0; e < 8; ++e) {
                float t = scale * acc[mi][ni][e] + bv;
                if (GELU)  t = 0.5f * t * (1.0f + erff(t * 0.70710678118654752f));
                if (RESID) t += R[base];
                C[base] = t;
                base += ldc;
            }
        }
}

// =====================================================================
// Row LayerNorm (one block per row); async row staging into LDS
// =====================================================================
__global__ __launch_bounds__(256)
void layernorm_k(const float* __restrict__ x, const float* __restrict__ w,
                 const float* __restrict__ b, float* __restrict__ y, int D)
{
    __shared__ float row_s[DM_];
    __shared__ float red[256];
    const long row = blockIdx.x;
    const float* xr = x + row * D;
    float* yr = y + row * D;

    if (threadIdx.x * 4 < D)
        async_ld_b128((unsigned)(size_t)&row_s[threadIdx.x * 4], xr + threadIdx.x * 4);
    wait_async0();
    __syncthreads();

    float s = 0.f;
    for (int i = threadIdx.x; i < D; i += 256) s += row_s[i];
    red[threadIdx.x] = s; __syncthreads();
    for (int o = 128; o > 0; o >>= 1) { if (threadIdx.x < o) red[threadIdx.x] += red[threadIdx.x + o]; __syncthreads(); }
    float mu = red[0] / (float)D; __syncthreads();
    float s2 = 0.f;
    for (int i = threadIdx.x; i < D; i += 256) { float d = row_s[i] - mu; s2 += d * d; }
    red[threadIdx.x] = s2; __syncthreads();
    for (int o = 128; o > 0; o >>= 1) { if (threadIdx.x < o) red[threadIdx.x] += red[threadIdx.x + o]; __syncthreads(); }
    float rinv = rsqrtf(red[0] / (float)D + 1e-6f);
    for (int i = threadIdx.x; i < D; i += 256) yr[i] = (row_s[i] - mu) * rinv * w[i] + b[i];
}

// =====================================================================
// Final RMSNorm -> d_out; async row staging
// =====================================================================
__global__ __launch_bounds__(256)
void rmsnorm_k(const float* __restrict__ x, const float* __restrict__ w,
               float* __restrict__ y, int D)
{
    __shared__ float row_s[DM_];
    __shared__ float red[256];
    const long row = blockIdx.x;
    const float* xr = x + row * D;
    float* yr = y + row * D;

    if (threadIdx.x * 4 < D)
        async_ld_b128((unsigned)(size_t)&row_s[threadIdx.x * 4], xr + threadIdx.x * 4);
    wait_async0();
    __syncthreads();

    float s2 = 0.f;
    for (int i = threadIdx.x; i < D; i += 256) s2 += row_s[i] * row_s[i];
    red[threadIdx.x] = s2; __syncthreads();
    for (int o = 128; o > 0; o >>= 1) { if (threadIdx.x < o) red[threadIdx.x] += red[threadIdx.x + o]; __syncthreads(); }
    float rinv = rsqrtf(red[0] / (float)D + 1e-6f);
    for (int i = threadIdx.x; i < D; i += 256) yr[i] = row_s[i] * rinv * w[i];
}

// =====================================================================
// Fused per-head LayerNorm (HD=64) + RoPE + transpose (b,s,h,d)->(b,h,s,d)
// one wave32 per (row, head); lane holds d and d+32
// =====================================================================
__global__ __launch_bounds__(128)
void hln_rope_k(const float* __restrict__ X, const float* __restrict__ gw,
                const float* __restrict__ gb, float* __restrict__ Y, int S)
{
    const int wid  = blockIdx.x * 4 + (threadIdx.x >> 5);  // (b*S+s)*NH + h
    const int lane = threadIdx.x & 31;
    const int h  = wid & (NH_ - 1);
    const int bs = wid >> 4;
    const int s  = bs % S;
    const int b  = bs / S;

    const float* xp = X + (long)wid * HD_;
    float x1 = xp[lane], x2 = xp[lane + 32];
    float sum = x1 + x2;
    for (int m = 16; m; m >>= 1) sum += __shfl_xor(sum, m, 32);
    float mu = sum * (1.f / 64.f);
    float d1 = x1 - mu, d2 = x2 - mu;
    float sq = d1 * d1 + d2 * d2;
    for (int m = 16; m; m >>= 1) sq += __shfl_xor(sq, m, 32);
    float rinv = rsqrtf(sq * (1.f / 64.f) + 1e-6f);
    float n1 = d1 * rinv * gw[lane]      + gb[lane];
    float n2 = d2 * rinv * gw[lane + 32] + gb[lane + 32];

    // RoPE: inv_freq = theta^(-lane/32); cos/sin identical for d and d+32
    float fr = (float)s * powf(10000.f, -(float)lane * (1.f / 32.f));
    float c = cosf(fr), sn = sinf(fr);
    float o1 = n1 * c - n2 * sn;          // d < 32 : x*c - x[d+32]*s
    float o2 = n2 * c + n1 * sn;          // d >= 32: x*c + x[d-32]*s

    float* yp = Y + ((long)(b * NH_ + h) * S + s) * HD_;
    yp[lane] = o1; yp[lane + 32] = o2;
}

// (b,s,h,d) -> (b,h,s,d)
__global__ __launch_bounds__(256)
void tr_bshd_to_bhsd(const float* __restrict__ in, float* __restrict__ out, int S)
{
    long i = (long)blockIdx.x * 256 + threadIdx.x;
    int d = (int)(i & 63); long t = i >> 6;
    int h = (int)(t & (NH_ - 1)); t >>= 4;
    int s = (int)(t % S); int b = (int)(t / S);
    out[(((long)(b * NH_ + h) * S + s) << 6) + d] = in[i];
}

// (b,h,s,d) -> (b,s,h,d)
__global__ __launch_bounds__(256)
void tr_bhsd_to_bshd(const float* __restrict__ in, float* __restrict__ out, int S)
{
    long i = (long)blockIdx.x * 256 + threadIdx.x;
    int d = (int)(i & 63); long t = i >> 6;
    int s = (int)(t % S); t /= S;
    int h = (int)(t & (NH_ - 1)); int b = (int)(t >> 4);
    out[((((long)b * S + s) * NH_ + h) << 6) + d] = in[i];
}

// in-place row softmax (width W<=1024): async row staging, 1 read + 1 write
__global__ __launch_bounds__(256)
void softmax_k(float* __restrict__ x, int W)
{
    __shared__ float row_s[SKV_];
    __shared__ float red[256];
    float* xr = x + (long)blockIdx.x * W;

    if (threadIdx.x * 4 < W)
        async_ld_b128((unsigned)(size_t)&row_s[threadIdx.x * 4], xr + threadIdx.x * 4);
    wait_async0();
    __syncthreads();

    float mx = -1e30f;
    for (int i = threadIdx.x; i < W; i += 256) mx = fmaxf(mx, row_s[i]);
    red[threadIdx.x] = mx; __syncthreads();
    for (int o = 128; o > 0; o >>= 1) { if (threadIdx.x < o) red[threadIdx.x] = fmaxf(red[threadIdx.x], red[threadIdx.x + o]); __syncthreads(); }
    float M = red[0]; __syncthreads();
    float s = 0.f;
    for (int i = threadIdx.x; i < W; i += 256) { float e = expf(row_s[i] - M); row_s[i] = e; s += e; }
    red[threadIdx.x] = s; __syncthreads();
    for (int o = 128; o > 0; o >>= 1) { if (threadIdx.x < o) red[threadIdx.x] += red[threadIdx.x + o]; __syncthreads(); }
    float inv = 1.f / red[0];
    for (int i = threadIdx.x; i < W; i += 256) xr[i] = row_s[i] * inv;
}

// embedding gather
__global__ __launch_bounds__(256)
void embed_k(const int* __restrict__ ids, const float* __restrict__ emb,
             float* __restrict__ out)
{
    const int row = blockIdx.x;
    const long id = (long)ids[row];
    for (int c = threadIdx.x; c < TGT_; c += 256)
        out[(long)row * TGT_ + c] = emb[id * TGT_ + c];
}

// =====================================================================
// Host-side orchestration
// =====================================================================
static inline const float* Fp(void* const* d_in, int i) { return (const float*)d_in[i]; }

// attention sub-graph (self or cross); p0: wq,wk,wv,wo,qn_w,qn_b,kn_w,kn_b
static void run_attn(hipStream_t stream, void* const* d_in, int p0,
                     const float* qin, const float* ctx, int ctxK, float* x,
                     float* qb, float* kb, float* vb,
                     float* qT, float* kT, float* vT, float* sc)
{
    dim3 blk(256);
    // Q/K/V projections (no bias in reference)
    gemm_wmma<true,false,false,false><<<dim3(DM_/64, ROWS_/128, 1), blk, 0, stream>>>(
        qin, DM_, 0, Fp(d_in,p0+0), DM_, 0, qb, DM_, 0, nullptr, nullptr, ROWS_, DM_, DM_, 1.f);
    gemm_wmma<true,false,false,false><<<dim3(DM_/64, ROWS_/128, 1), blk, 0, stream>>>(
        ctx, ctxK, 0, Fp(d_in,p0+1), ctxK, 0, kb, DM_, 0, nullptr, nullptr, ROWS_, DM_, ctxK, 1.f);
    gemm_wmma<true,false,false,false><<<dim3(DM_/64, ROWS_/128, 1), blk, 0, stream>>>(
        ctx, ctxK, 0, Fp(d_in,p0+2), ctxK, 0, vb, DM_, 0, nullptr, nullptr, ROWS_, DM_, ctxK, 1.f);

    // head-LN + RoPE + transpose to (b,h,s,d)
    hln_rope_k<<<16384, 128, 0, stream>>>(qb, Fp(d_in,p0+4), Fp(d_in,p0+5), qT, SQ_);
    hln_rope_k<<<16384, 128, 0, stream>>>(kb, Fp(d_in,p0+6), Fp(d_in,p0+7), kT, SKV_);
    tr_bshd_to_bhsd<<<16384, 256, 0, stream>>>(vb, vT, SKV_);

    // per-batch-index attention: scores -> softmax -> P@V  (16 heads batched)
    // 64 MB score chunk stays L2-resident (192 MB L2)
    const long HS = (long)SQ_ * HD_;            // 65536 per (b,h)
    for (int b = 0; b < B_; ++b) {
        const float* qTb = qT + (long)b * NH_ * HS;
        const float* kTb = kT + (long)b * NH_ * HS;
        const float* vTb = vT + (long)b * NH_ * HS;
        float*       ob  = qb + (long)b * NH_ * HS;    // reuse qb as O (b,h,s,d)
        gemm_wmma<true,false,false,false><<<dim3(SKV_/64, SQ_/128, NH_), blk, 0, stream>>>(
            qTb, HD_, HS, kTb, HD_, HS, sc, SKV_, (long)SQ_*SKV_,
            nullptr, nullptr, SQ_, SKV_, HD_, 0.125f);
        softmax_k<<<NH_ * SQ_, 256, 0, stream>>>(sc, SKV_);
        gemm_wmma<false,false,false,false><<<dim3(HD_/64, SQ_/128, NH_), blk, 0, stream>>>(
            sc, SKV_, (long)SQ_*SKV_, vTb, HD_, HS, ob, HD_, HS,
            nullptr, nullptr, SQ_, HD_, SKV_, 1.f);
    }

    // merge heads (b,h,s,d)->(b,s,h*d) then output proj with residual into x
    tr_bhsd_to_bshd<<<16384, 256, 0, stream>>>(qb, kb, SQ_);
    gemm_wmma<true,false,false,true><<<dim3(DM_/64, ROWS_/128, 1), blk, 0, stream>>>(
        kb, DM_, 0, Fp(d_in,p0+3), DM_, 0, x, DM_, 0, nullptr, x, ROWS_, DM_, DM_, 1.f);
}

extern "C" void kernel_launch(void* const* d_in, const int* in_sizes, int n_in,
                              void* d_out, int out_size, void* d_ws, size_t ws_size,
                              hipStream_t stream)
{
    (void)in_sizes; (void)n_in; (void)out_size; (void)ws_size;
    const float* src   = Fp(d_in, 0);
    const int*   ids   = (const int*)d_in[1];
    const float* embed = Fp(d_in, 2);
    const float* in_w  = Fp(d_in, 3);
    const float* in_b  = Fp(d_in, 4);
    const float* out_w = Fp(d_in, 5);
    const float* out_b = Fp(d_in, 6);
    const float* fnw   = Fp(d_in, 7);
    // per-block leaves: [0]ln_sa_w [1]ln_sa_b [2..9]sa{wq,wk,wv,wo,qn_w,qn_b,kn_w,kn_b}
    //                   [10]ln_ca_w [11]ln_ca_b [12..19]ca{...}
    //                   [20]ln_mlp_w [21]ln_mlp_b [22]w1 [23]b1 [24]w2 [25]b2

    float* ws = (float*)d_ws;
    const long SZ = (long)ROWS_ * DM_;          // 4,194,304 floats
    float* x  = ws;
    float* nb = x  + SZ;
    float* qb = nb + SZ;
    float* kb = qb + SZ;
    float* vb = kb + SZ;
    float* qT = vb + SZ;
    float* kT = qT + SZ;
    float* vT = kT + SZ;
    float* sc = vT + SZ;                        // 16,777,216 floats: scores(per-b) / MLP hidden
    float* x0 = sc + (long)NH_ * SQ_ * SKV_;
    float* yb = x0 + (long)ROWS_ * TGT_;

    dim3 blk(256);

    // embedding + input projection
    embed_k<<<ROWS_, 256, 0, stream>>>(ids, embed, x0);
    gemm_wmma<true,true,false,false><<<dim3(DM_/64, ROWS_/128, 1), blk, 0, stream>>>(
        x0, TGT_, 0, in_w, TGT_, 0, x, DM_, 0, in_b, nullptr, ROWS_, DM_, TGT_, 1.f);

    for (int L = 0; L < 4; ++L) {
        const int pb = 8 + L * 26;
        // self-attention
        layernorm_k<<<ROWS_, 256, 0, stream>>>(x, Fp(d_in,pb+0), Fp(d_in,pb+1), nb, DM_);
        run_attn(stream, d_in, pb + 2, nb, nb, DM_, x, qb, kb, vb, qT, kT, vT, sc);
        // cross-attention
        layernorm_k<<<ROWS_, 256, 0, stream>>>(x, Fp(d_in,pb+10), Fp(d_in,pb+11), nb, DM_);
        run_attn(stream, d_in, pb + 12, nb, src, SRC_, x, qb, kb, vb, qT, kT, vT, sc);
        // MLP
        layernorm_k<<<ROWS_, 256, 0, stream>>>(x, Fp(d_in,pb+20), Fp(d_in,pb+21), nb, DM_);
        gemm_wmma<true,true,true,false><<<dim3(DFF_/64, ROWS_/128, 1), blk, 0, stream>>>(
            nb, DM_, 0, Fp(d_in,pb+22), DM_, 0, sc, DFF_, 0, Fp(d_in,pb+23), nullptr,
            ROWS_, DFF_, DM_, 1.f);
        gemm_wmma<true,true,false,true><<<dim3(DM_/64, ROWS_/128, 1), blk, 0, stream>>>(
            sc, DFF_, 0, Fp(d_in,pb+24), DFF_, 0, x, DM_, 0, Fp(d_in,pb+25), x,
            ROWS_, DM_, DFF_, 1.f);
    }

    // output projection + final RMSNorm
    gemm_wmma<true,true,false,false><<<dim3(TGT_/64, ROWS_/128, 1), blk, 0, stream>>>(
        x, DM_, 0, out_w, DM_, 0, yb, TGT_, 0, out_b, nullptr, ROWS_, TGT_, DM_, 1.f);
    rmsnorm_k<<<ROWS_, 256, 0, stream>>>(yb, fnw, (float*)d_out, TGT_);
}
